// p_net_7559142441301
// MI455X (gfx1250) — compile-verified, matrix operand order checked by reading, not run on hardware
//
#include <hip/hip_runtime.h>
#include <hip/hip_bf16.h>
#include <math.h>

typedef __attribute__((ext_vector_type(16))) __bf16 v16bf;
typedef __attribute__((ext_vector_type(8)))  __bf16 v8bf;
typedef __attribute__((ext_vector_type(8)))  float  v8f;
typedef __attribute__((ext_vector_type(4)))  float  v4f;

#define BB 4096

// ---- workspace (prepacked bf16 weights) element offsets ----
#define W2P_OFF 0        // [9][128][64]  = 73728
#define W3P_OFF 73728    // [9][128][128] = 147456
#define INP_OFF 221184   // [384][128]    = 49152
#define OWP_OFF 270336   // [128][128]    = 16384
#define L1P_OFF 286720   // [128][128]    = 16384
#define L2P_OFF 303104   // [128][128]    = 16384  -> total 319488 elems (639 KB)

// ---- dynamic LDS layout (bytes) ----
#define OFF_XT    0        // 96*128 bf16 = 24576  (xT / xln)
#define OFF_QK    24576    // 96*256 bf16 = 49152  (q,k rows; reused as xpre f32 [96*128])
#define OFF_VT    73728    // 128*96 bf16 = 24576  (v transposed: [d][token])
#define OFF_PROBS 98304    // 96*96  bf16 = 18432  (probs; reused as x1t conv buf [121][64])
#define OFF_ASC   116736   // 96*96  f32  = 36864  (asc accum; reused as x2t conv buf [121][128])
#define OFF_CTX   153600   // 96*128 bf16 = 24576  (ctx / y1)
#define OFF_SC    178176   // 96*96  f32  = 36864  (raw scores; reused as y2 bf16)
#define OFF_DIG   215040
#define OFF_MISC  215424
#define SMEM_BYTES 215936

__device__ __forceinline__ v8f wmma_bf16(v16bf a, v16bf b, v8f c) {
  return __builtin_amdgcn_wmma_f32_16x16x32_bf16(false, a, false, b, (short)0, c,
                                                 false, false);
}

// Fragment load per CDNA5 wave32 16-bit layout: two contiguous 16B runs.
// row = pointer to this lane's matrix row (A: row r, B: col c), kb = K base.
__device__ __forceinline__ v16bf frag_ld(const __bf16* row, int kb) {
  const int g = (threadIdx.x >> 4) & 1;
  const v8bf lo = *(const v8bf*)(row + kb + g * 8);
  const v8bf hi = *(const v8bf*)(row + kb + 16 + g * 8);
  return __builtin_shufflevector(lo, hi, 0, 1, 2, 3, 4, 5, 6, 7,
                                 8, 9, 10, 11, 12, 13, 14, 15);
}

__device__ __forceinline__ float siluf(float x) { return x / (1.0f + __expf(-x)); }

__device__ __forceinline__ float attn_mask(int h, int m, int n) {
  const int r1 = m / 9, c1 = m % 9, r2 = n / 9, c2 = n % 9;
  bool same;
  if (h == 0) same = (r1 == r2);
  else if (h == 1) same = (c1 == c2);
  else if (h == 2) same = ((r1 / 3) == (r2 / 3)) && ((c1 / 3) == (c2 / 3));
  else same = true;
  return same ? 1.0f : 0.0f;
}

__device__ __forceinline__ unsigned hashu(unsigned x) {
  x ^= x >> 16; x *= 0x7feb352du; x ^= x >> 15; x *= 0x846ca68bu; x ^= x >> 16;
  return x;
}
__device__ __forceinline__ float u01(unsigned x) {
  return (hashu(x) >> 8) * (1.0f / 16777216.0f) + 1e-12f;
}

// ---- one-time weight conversion/repack: f32 -> bf16, conv taps split out ----
__global__ void prepack_weights(const float* __restrict__ c2w, const float* __restrict__ c3w,
                                const float* __restrict__ inw, const float* __restrict__ ow,
                                const float* __restrict__ l1w, const float* __restrict__ l2w,
                                __bf16* __restrict__ ws) {
  const int t0 = blockIdx.x * blockDim.x + threadIdx.x;
  const int stp = gridDim.x * blockDim.x;
  for (int i = t0; i < 9 * 128 * 64; i += stp) {        // w2p[tap][o][ci]
    const int tap = i >> 13, o = (i >> 6) & 127, ci = i & 63;
    ws[W2P_OFF + i] = (__bf16)c2w[o * 576 + ci * 9 + tap];
  }
  for (int i = t0; i < 9 * 128 * 128; i += stp) {       // w3p[tap][o][ci]
    const int tap = i >> 14, o = (i >> 7) & 127, ci = i & 127;
    ws[W3P_OFF + i] = (__bf16)c3w[o * 1152 + ci * 9 + tap];
  }
  for (int i = t0; i < 384 * 128; i += stp) ws[INP_OFF + i] = (__bf16)inw[i];
  for (int i = t0; i < 128 * 128; i += stp) {
    ws[OWP_OFF + i] = (__bf16)ow[i];
    ws[L1P_OFF + i] = (__bf16)l1w[i];
    ws[L2P_OFF + i] = (__bf16)l2w[i];
  }
}

__global__ void sudoku_fused_kernel(
    const float* __restrict__ s,
    const float* __restrict__ c1w, const float* __restrict__ c1b,
    const float* __restrict__ c2b, const float* __restrict__ c3b,
    const float* __restrict__ emb, const float* __restrict__ inb,
    const float* __restrict__ obias,
    const float* __restrict__ lng, const float* __restrict__ lnb,
    const float* __restrict__ l1b, const float* __restrict__ l2b,
    const float* __restrict__ pw,  const float* __restrict__ pb,
    const float* __restrict__ nw,  const float* __restrict__ nb,
    const __bf16* __restrict__ ws,
    float* __restrict__ out) {
  extern __shared__ __align__(16) char smem[];
  const int b    = blockIdx.x;
  const int tid  = threadIdx.x;
  const int wave = tid >> 5;
  const int lane = tid & 31;
  const int g    = lane >> 4;
  const int cn   = lane & 15;

  __bf16* xT    = (__bf16*)(smem + OFF_XT);     // [96][128]
  __bf16* qk    = (__bf16*)(smem + OFF_QK);     // [96][256]
  __bf16* vT    = (__bf16*)(smem + OFF_VT);     // [128][96]
  __bf16* probs = (__bf16*)(smem + OFF_PROBS);  // [96][96]
  float*  ascA  = (float*)(smem + OFF_ASC);     // [96][96]
  __bf16* ctx   = (__bf16*)(smem + OFF_CTX);    // [96][128]
  float*  sc    = (float*)(smem + OFF_SC);      // [96][96]
  int*    dig   = (int*)(smem + OFF_DIG);
  float*  plog  = (float*)(smem + OFF_MISC);
  float*  nlog  = plog + 96;
  int*    ipos  = (int*)(nlog + 16);
  __bf16* x1t = (__bf16*)(smem + OFF_PROBS);    // [121][64]  spatial-major, padded
  __bf16* x2t = (__bf16*)(smem + OFF_ASC);      // [121][128] spatial-major, padded

  const __bf16* wsW2 = ws + W2P_OFF;
  const __bf16* wsW3 = ws + W3P_OFF;
  const __bf16* wsIN = ws + INP_OFF;
  const __bf16* wsOW = ws + OWP_OFF;
  const __bf16* wsL1 = ws + L1P_OFF;
  const __bf16* wsL2 = ws + L2P_OFF;

  if (tid < 16) {  // global_prefetch_b8: warm L2 for repacked conv weights
    __builtin_prefetch(wsW2 + tid * 4608, 0, 0);
    __builtin_prefetch(wsW3 + tid * 9216, 0, 0);
  }

  // ---- phase 0: zero padded conv bufs + xT rows, extract given digits ----
  {
    const v4f z4 = {0.0f, 0.0f, 0.0f, 0.0f};
    v4f* za = (v4f*)x1t; v4f* zb = (v4f*)x2t; v4f* zc = (v4f*)xT;
    for (int i = tid; i < 968;  i += 256) za[i] = z4;
    for (int i = tid; i < 1936; i += 256) zb[i] = z4;
    for (int i = tid; i < 1536; i += 256) zc[i] = z4;
  }
  if (tid < 81) {
    const float* sp = s + (size_t)b * 810 + tid;
    int d = 0;
    for (int i = 1; i < 10; ++i) if (sp[i * 81] > 0.5f) d = i;
    dig[tid] = d;
  }
  __syncthreads();

  // ---- 1x1 conv via one-hot gather -> x1t[sp][ci], packed stores ----
  for (int t = tid; t < 81 * 8; t += 256) {
    const int p = t >> 3, c0 = (t & 7) * 8;
    const int dg = dig[p];
    const int sp = (p / 9 + 1) * 11 + (p % 9 + 1);
    v8bf o;
#pragma unroll
    for (int q = 0; q < 8; ++q) o[q] = (__bf16)(c1w[(c0 + q) * 10 + dg] + c1b[c0 + q]);
    *(v8bf*)(x1t + sp * 64 + c0) = o;
  }
  __syncthreads();

  // per-lane token geometry for the conv N-tiles
  int spb[6];
#pragma unroll
  for (int nt = 0; nt < 6; ++nt) {
    int n = nt * 16 + cn; if (n > 80) n = 80;
    spb[nt] = (n / 9) * 11 + (n % 9);
  }

  // ---- conv2 as 9 shifted GEMMs: [128 x 64] @ [64 x 81] per tap, silu -> x2t ----
  {
    const int mt = wave;                       // 8 waves = 128 output channels
    v8f acc[6] = {};
    for (int tap = 0; tap < 9; ++tap) {
      const int offadd = (tap / 3) * 11 + (tap % 3);
      const __bf16* arow = wsW2 + tap * 8192 + (mt * 16 + cn) * 64;
#pragma unroll
      for (int kc = 0; kc < 2; ++kc) {
        const v16bf a = frag_ld(arow, kc * 32);
#pragma unroll
        for (int nt = 0; nt < 6; ++nt) {
          const v16bf bf = frag_ld(x1t + (spb[nt] + offadd) * 64, kc * 32);
          acc[nt] = wmma_bf16(a, bf, acc[nt]);
        }
      }
    }
    const int mb = mt * 16 + g * 8;
    const v4f b0 = *(const v4f*)(c2b + mb);
    const v4f b1 = *(const v4f*)(c2b + mb + 4);
#pragma unroll
    for (int nt = 0; nt < 6; ++nt) {
      if (nt * 16 + cn < 81) {
        v8bf o;
#pragma unroll
        for (int r = 0; r < 8; ++r)
          o[r] = (__bf16)siluf(acc[nt][r] + (r < 4 ? b0[r] : b1[r - 4]));
        *(v8bf*)(x2t + (spb[nt] + 12) * 128 + mb) = o;
      }
    }
  }
  __syncthreads();

  // ---- conv3: 9 shifted GEMMs [128 x 128] @ [128 x 81], silu, +emb -> xT[n][d] ----
  {
    const int mt = wave;
    v8f acc[6] = {};
    for (int tap = 0; tap < 9; ++tap) {
      const int offadd = (tap / 3) * 11 + (tap % 3);
      const __bf16* arow = wsW3 + tap * 16384 + (mt * 16 + cn) * 128;
#pragma unroll
      for (int kc = 0; kc < 4; ++kc) {
        const v16bf a = frag_ld(arow, kc * 32);
#pragma unroll
        for (int nt = 0; nt < 6; ++nt) {
          const v16bf bf = frag_ld(x2t + (spb[nt] + offadd) * 128, kc * 32);
          acc[nt] = wmma_bf16(a, bf, acc[nt]);
        }
      }
    }
    const int mb = mt * 16 + g * 8;
    const v4f b0 = *(const v4f*)(c3b + mb);
    const v4f b1 = *(const v4f*)(c3b + mb + 4);
#pragma unroll
    for (int nt = 0; nt < 6; ++nt) {
      const int n = nt * 16 + cn;
      if (n < 81) {
        const v4f e0 = *(const v4f*)(emb + n * 128 + mb);
        const v4f e1 = *(const v4f*)(emb + n * 128 + mb + 4);
        v8bf o;
#pragma unroll
        for (int r = 0; r < 8; ++r) {
          const float bb = (r < 4 ? b0[r] : b1[r - 4]);
          const float ee = (r < 4 ? e0[r] : e1[r - 4]);
          o[r] = (__bf16)(siluf(acc[nt][r] + bb) + ee);
        }
        *(v8bf*)(xT + n * 128 + mb) = o;
      }
    }
  }
  __syncthreads();

  // ---- qkv: M = feature j (24 tiles), N = token (6 tiles), K = 128 ----
  for (int t = wave; t < 144; t += 8) {
    const int mt = t / 6, nt = t % 6;
    const __bf16* arow = wsIN + (mt * 16 + cn) * 128;
    const __bf16* brow = xT + (nt * 16 + cn) * 128;
    v8f acc = {};
#pragma unroll
    for (int kc = 0; kc < 4; ++kc)
      acc = wmma_bf16(frag_ld(arow, kc * 32), frag_ld(brow, kc * 32), acc);
    const int ntok = nt * 16 + cn;
    const int jb = mt * 16 + g * 8;
    const v4f b0 = *(const v4f*)(inb + jb);
    const v4f b1 = *(const v4f*)(inb + jb + 4);
    if (jb < 256) {                       // q,k rows: contiguous b128 store
      v8bf o;
#pragma unroll
      for (int r = 0; r < 8; ++r)
        o[r] = (__bf16)(acc[r] + (r < 4 ? b0[r] : b1[r - 4]));
      *(v8bf*)(qk + ntok * 256 + jb) = o;
    } else {                              // v: transpose into vT[d][token]
#pragma unroll
      for (int r = 0; r < 8; ++r)
        vT[(jb + r - 256) * 96 + ntok] = (__bf16)(acc[r] + (r < 4 ? b0[r] : b1[r - 4]));
    }
  }
  for (int i = tid; i < 96 * 96; i += 256) ascA[i] = 0.0f;
  __syncthreads();

  // ---- attention heads ----
  for (int h = 0; h < 4; ++h) {
    // scores = q k^T / sqrt(32) + structural mask (K=32: single WMMA per tile)
    for (int t = wave; t < 36; t += 8) {
      const int mt = t / 6, nt = t % 6;
      const __bf16* arow = qk + (mt * 16 + cn) * 256 + h * 32;
      const __bf16* brow = qk + (nt * 16 + cn) * 256 + 128 + h * 32;
      v8f acc = {};
      acc = wmma_bf16(frag_ld(arow, 0), frag_ld(brow, 0), acc);
      const int n = nt * 16 + cn;
#pragma unroll
      for (int r = 0; r < 8; ++r) {
        const int m = mt * 16 + r + g * 8;
        float v = acc[r] * 0.17677669529663687f;
        v = (m < 81 && n < 81) ? (v + attn_mask(h, m, n)) : -1e30f;
        sc[m * 96 + n] = v;
      }
    }
    __syncthreads();
    // row softmax + asc accumulation, packed prob stores
    if (tid < 96) {
      const int m = tid;
      float mx = -1e30f;
      for (int n = 0; n < 96; ++n) mx = fmaxf(mx, sc[m * 96 + n]);
      float sum = 0.0f;
      for (int n = 0; n < 96; ++n) sum += __expf(sc[m * 96 + n] - mx);
      const float inv = 1.0f / sum;
      for (int n8 = 0; n8 < 96; n8 += 8) {
        v8bf o;
#pragma unroll
        for (int q = 0; q < 8; ++q) {
          const float p = __expf(sc[m * 96 + n8 + q] - mx) * inv;
          o[q] = (__bf16)p;
          ascA[m * 96 + n8 + q] += 0.25f * p;
        }
        *(v8bf*)(probs + m * 96 + n8) = o;
      }
    }
    __syncthreads();
    // ctx[:, h*32:(h+1)*32] = probs @ v_h  (B rows from vT: K=token contiguous)
    for (int t = wave; t < 12; t += 8) {
      const int mt = t % 6, ntl = t / 6;
      const __bf16* arow = probs + (mt * 16 + cn) * 96;
      const __bf16* brow = vT + (h * 32 + ntl * 16 + cn) * 96;
      v8f acc = {};
#pragma unroll
      for (int kc = 0; kc < 3; ++kc)
        acc = wmma_bf16(frag_ld(arow, kc * 32), frag_ld(brow, kc * 32), acc);
      const int d = h * 32 + ntl * 16 + cn;
#pragma unroll
      for (int r = 0; r < 8; ++r)
        ctx[(mt * 16 + r + g * 8) * 128 + d] = (__bf16)acc[r];
    }
    __syncthreads();
  }

  // ---- out projection: M = feature j, B = ctx rows -> xpre f32 [n][128] ----
  float* xpre = (float*)(smem + OFF_QK);
  for (int t = wave; t < 48; t += 8) {
    const int mt = t / 6, nt = t % 6;
    const __bf16* arow = wsOW + (mt * 16 + cn) * 128;
    const __bf16* brow = ctx + (nt * 16 + cn) * 128;
    v8f acc = {};
#pragma unroll
    for (int kc = 0; kc < 4; ++kc)
      acc = wmma_bf16(frag_ld(arow, kc * 32), frag_ld(brow, kc * 32), acc);
    const int ntok = nt * 16 + cn;
    const int jb = mt * 16 + g * 8;
    const v4f b0 = *(const v4f*)(obias + jb);
    const v4f b1 = *(const v4f*)(obias + jb + 4);
    v4f lo, hi;
#pragma unroll
    for (int r = 0; r < 4; ++r) { lo[r] = acc[r] + b0[r]; hi[r] = acc[r + 4] + b1[r]; }
    *(v4f*)(xpre + ntok * 128 + jb) = lo;
    *(v4f*)(xpre + ntok * 128 + jb + 4) = hi;
  }
  __syncthreads();

  // ---- LayerNorm -> xln (xT region), packed stores ----
  __bf16* xln = xT;
  if (tid < 96) {
    const int m = tid;
    float mu = 0.0f;
    for (int d = 0; d < 128; ++d) mu += xpre[m * 128 + d];
    mu *= (1.0f / 128.0f);
    float var = 0.0f;
    for (int d = 0; d < 128; ++d) { const float t2 = xpre[m * 128 + d] - mu; var += t2 * t2; }
    const float inv = rsqrtf(var * (1.0f / 128.0f) + 1e-5f);
    for (int d8 = 0; d8 < 128; d8 += 8) {
      v8bf o;
#pragma unroll
      for (int q = 0; q < 8; ++q)
        o[q] = (__bf16)((xpre[m * 128 + d8 + q] - mu) * inv * lng[d8 + q] + lnb[d8 + q]);
      *(v8bf*)(xln + m * 128 + d8) = o;
    }
  }
  __syncthreads();

  // ---- l1: silu(xln @ l1w^T + l1b) -> y1 (ctx region) ----
  __bf16* y1 = ctx;
  for (int t = wave; t < 48; t += 8) {
    const int mt = t / 6, nt = t % 6;
    const __bf16* arow = wsL1 + (mt * 16 + cn) * 128;
    const __bf16* brow = xln + (nt * 16 + cn) * 128;
    v8f acc = {};
#pragma unroll
    for (int kc = 0; kc < 4; ++kc)
      acc = wmma_bf16(frag_ld(arow, kc * 32), frag_ld(brow, kc * 32), acc);
    const int ntok = nt * 16 + cn;
    const int jb = mt * 16 + g * 8;
    const v4f b0 = *(const v4f*)(l1b + jb);
    const v4f b1 = *(const v4f*)(l1b + jb + 4);
    v8bf o;
#pragma unroll
    for (int r = 0; r < 8; ++r)
      o[r] = (__bf16)siluf(acc[r] + (r < 4 ? b0[r] : b1[r - 4]));
    *(v8bf*)(y1 + ntok * 128 + jb) = o;
  }
  __syncthreads();

  // ---- l2: silu(y1 @ l2w^T + l2b) -> y2 (sc region as bf16) ----
  __bf16* y2 = (__bf16*)(smem + OFF_SC);
  for (int t = wave; t < 48; t += 8) {
    const int mt = t / 6, nt = t % 6;
    const __bf16* arow = wsL2 + (mt * 16 + cn) * 128;
    const __bf16* brow = y1 + (nt * 16 + cn) * 128;
    v8f acc = {};
#pragma unroll
    for (int kc = 0; kc < 4; ++kc)
      acc = wmma_bf16(frag_ld(arow, kc * 32), frag_ld(brow, kc * 32), acc);
    const int ntok = nt * 16 + cn;
    const int jb = mt * 16 + g * 8;
    const v4f b0 = *(const v4f*)(l2b + jb);
    const v4f b1 = *(const v4f*)(l2b + jb + 4);
    v8bf o;
#pragma unroll
    for (int r = 0; r < 8; ++r)
      o[r] = (__bf16)siluf(acc[r] + (r < 4 ? b0[r] : b1[r - 4]));
    *(v8bf*)(y2 + ntok * 128 + jb) = o;
  }
  __syncthreads();

  // ---- position head (masked) + deterministic Gumbel sampling ----
  if (tid < 81) {
    float v = pb[0];
    for (int d = 0; d < 128; ++d) v += (float)y2[tid * 128 + d] * pw[d];
    if (dig[tid] != 0) v = -1e9f;
    plog[tid] = v;
  }
  __syncthreads();
  if (tid == 0) {
    int best = 0; float bv = -1e30f;
    for (int i = 0; i < 81; ++i) {
      const float gmb = -__logf(-__logf(u01((unsigned)b * 131071u + 0x9e3779b9u + (unsigned)i)));
      const float v = plog[i] + gmb;
      if (v > bv) { bv = v; best = i; }
    }
    *ipos = best;
    out[b] = (float)best;
  }
  __syncthreads();
  const int pos = *ipos;
  if (tid < 10) {
    float v = nb[tid];
    for (int d = 0; d < 128; ++d) v += (float)y2[pos * 128 + d] * nw[tid * 128 + d];
    nlog[tid] = (tid == 0) ? -1e30f : v;
  }
  __syncthreads();
  if (tid == 0) {
    int best = 1; float bv = -1e30f;
    for (int i = 1; i < 10; ++i) {
      const float gmb = -__logf(-__logf(u01((unsigned)b * 524287u + 0x85ebca6bu + (unsigned)i)));
      const float v = nlog[i] + gmb;
      if (v > bv) { bv = v; best = i; }
    }
    out[BB + b] = (float)best;
  }

  // ---- stream asc out (108 MB): non-temporal dword stores, no div/mod in loop ----
  // 243 threads cover rows {m0, m0+3, ..., m0+78}; one /81 per thread total.
  if (tid < 243) {
    float* ascout = out + 2 * BB + (size_t)b * 6561;
    const int m0 = tid / 81, n0 = tid - m0 * 81;
    int m = m0;
#pragma unroll 1
    for (int pass = 0; pass < 27; ++pass, m += 3)
      __builtin_nontemporal_store(ascA[m * 96 + n0], ascout + m * 81 + n0);
  }
}

extern "C" void kernel_launch(void* const* d_in, const int* in_sizes, int n_in,
                              void* d_out, int out_size, void* d_ws, size_t ws_size,
                              hipStream_t stream) {
  (void)in_sizes; (void)n_in; (void)out_size; (void)ws_size;
  const float* s   = (const float*)d_in[0];
  const float* c1w = (const float*)d_in[1];
  const float* c1b = (const float*)d_in[2];
  const float* c2w = (const float*)d_in[3];
  const float* c2b = (const float*)d_in[4];
  const float* c3w = (const float*)d_in[5];
  const float* c3b = (const float*)d_in[6];
  const float* emb = (const float*)d_in[7];
  const float* inw = (const float*)d_in[8];
  const float* inb = (const float*)d_in[9];
  const float* ow  = (const float*)d_in[10];
  const float* ob  = (const float*)d_in[11];
  const float* lng = (const float*)d_in[12];
  const float* lnb = (const float*)d_in[13];
  const float* l1w = (const float*)d_in[14];
  const float* l1b = (const float*)d_in[15];
  const float* l2w = (const float*)d_in[16];
  const float* l2b = (const float*)d_in[17];
  const float* pw  = (const float*)d_in[18];
  const float* pb  = (const float*)d_in[19];
  const float* nw  = (const float*)d_in[20];
  const float* nb  = (const float*)d_in[21];
  float* out = (float*)d_out;
  __bf16* ws = (__bf16*)d_ws;

  prepack_weights<<<312, 256, 0, stream>>>(c2w, c3w, inw, ow, l1w, l2w, ws);

  hipFuncSetAttribute(reinterpret_cast<const void*>(sudoku_fused_kernel),
                      hipFuncAttributeMaxDynamicSharedMemorySize, SMEM_BYTES);
  sudoku_fused_kernel<<<BB, 256, SMEM_BYTES, stream>>>(
      s, c1w, c1b, c2b, c3b, emb, inb, ob, lng, lnb, l1b, l2b,
      pw, pb, nw, nb, ws, out);
}
